// InputModule_43198781063642
// MI455X (gfx1250) — matrix operand order repaired and checked by reading, not compiled, not run on hardware
//
#include <hip/hip_runtime.h>
#include <hip/hip_bf16.h>

// ---------------- problem constants ----------------
constexpr int BATCH = 64;
constexpr int SEQN  = 1024;
constexpr int EMBD  = 256;
constexpr int HIDN  = 512;
constexpr int GATE3 = 3 * HIDN;   // 1536
constexpr int EOS   = 2;

// ---------------- kernel config ----------------
constexpr int NWG = 16;           // persistent workgroups (grid barrier participants)
constexpr int HS  = HIDN / NWG;   // 32 hidden columns per WG
constexpr int NT  = 384;          // 12 waves; wave -> (ni = w>>1 in 0..5, mih = w&1)
constexpr int PF  = 2;            // A-fragment staging depth (LDS latency is short)

constexpr int HSTRIDE = BATCH * HIDN;   // elements per bf16 h buffer

// ---- LDS slab geometry (rows padded by 16B so ds_load_b128 spreads banks) ----
constexpr int HROW_U4 = 65;                       // 64 data u4 + 1 pad u4 per h row
constexpr int EROW_U4 = 33;                       // 32 data u4 + 1 pad u4 per emb row
constexpr int HROW_E  = HROW_U4 * 8;              // 520 bf16 per padded h row
constexpr int EROW_E  = EROW_U4 * 8;              // 264 bf16 per padded emb row
constexpr size_t HLDS_BYTES = (size_t)BATCH * HROW_U4 * 16;   // 66560
constexpr size_t ELDS_BYTES = (size_t)BATCH * EROW_U4 * 16;   // 33792
constexpr size_t DYN_BYTES  = HLDS_BYTES + 2 * ELDS_BYTES;    // 134144

// ---------------- workspace layout (bytes) ----------------
constexpr size_t OFF_BAR  = 0;
constexpr size_t OFF_HBF  = 256;                                  // 2 x [64][512] bf16 ping-pong
constexpr size_t OFF_DEST = OFF_HBF + 2 * (size_t)HSTRIDE * 2;
constexpr size_t OFF_WIH  = OFF_DEST + (size_t)SEQN * BATCH * 4;
constexpr size_t OFF_WHH  = OFF_WIH + (size_t)GATE3 * EMBD * 2;
constexpr size_t OFF_EMBB = OFF_WHH + (size_t)GATE3 * HIDN * 2;
constexpr size_t WS_NEED  = OFF_EMBB + (size_t)SEQN * BATCH * EMBD * 2;

typedef __attribute__((ext_vector_type(16))) __bf16 v16bf;
typedef __attribute__((ext_vector_type(8)))  float  v8f;
typedef __attribute__((ext_vector_type(4)))  unsigned v4u;
typedef __attribute__((ext_vector_type(8)))  int   v8i_;
typedef __attribute__((ext_vector_type(4)))  int   v4i_;

#define HAVE_TDM __has_builtin(__builtin_amdgcn_tensor_load_to_lds)

union Frag32B { uint4 u[2]; v16bf v; };

// Unused helper (B loads use load_frag_split): per-lane 16 contiguous bf16 (32 bytes)
__device__ __forceinline__ v16bf load_frag_contig(const __bf16* p) {
  Frag32B t;
  t.u[0] = *(const uint4*)(p);
  t.u[1] = *(const uint4*)(p + 8);
  return t.v;
}
// A fragment: per-lane two runs of 8 bf16, 16 elements apart (ISA 16-bit A layout)
__device__ __forceinline__ v16bf load_frag_split(const __bf16* p) {
  Frag32B t;
  t.u[0] = *(const uint4*)(p);
  t.u[1] = *(const uint4*)(p + 16);
  return t.v;
}

__device__ __forceinline__ v8f vzero8() {
  v8f v;
#pragma unroll
  for (int i = 0; i < 8; ++i) v[i] = 0.0f;
  return v;
}

__device__ __forceinline__ float fast_sigmoid(float x) {
  return __builtin_amdgcn_rcpf(1.0f + __expf(-x));
}
__device__ __forceinline__ float fast_tanh(float x) {
  float e2 = __expf(2.0f * x);
  return (e2 - 1.0f) * __builtin_amdgcn_rcpf(e2 + 1.0f);
}

#if HAVE_TDM
// One-row 2B-element TDM transfer of `nelem` elements global->LDS, inserting
// LDS padding every (1<<pad_code)*2 DWORDs (pad_amount_m1+1 DWORDs each).
// D# bit layout per CDNA5 ISA ch.8. Generic LDS pointer low 32 bits == LDS
// byte address (ISA 10.2 aperture rule). 6-arg builtin form (clang-23):
// (uint32x4 g0, int32x8 g1, int32x4 g2, int32x4 g3, int32x8, i32 cpol).
__device__ __forceinline__ void tdm_load_1d(const void* gsrc, unsigned lds_byte,
                                            unsigned nelem, unsigned pad_code,
                                            unsigned pad_amount_m1) {
  unsigned long long ga = (unsigned long long)gsrc;
  v4u g0;
  g0[0] = 1u;                                                   // count=1 user D#
  g0[1] = lds_byte;                                             // lds_addr
  g0[2] = (unsigned)ga;                                         // global_addr[31:0]
  g0[3] = (unsigned)((ga >> 32) & 0x01FFFFFFu) | (2u << 30);    // addr[56:32] | type=2
  v8i_ g1;
  g1[0] = (int)((1u << 16) |                                    // data_size = 2 bytes
                (1u << 20) |                                    // pad_enable
                (pad_code << 22) | (pad_amount_m1 << 25));      // pad interval/amount
  g1[1] = (int)((nelem & 0xFFFFu) << 16);                       // tensor_dim0[15:0]
  g1[2] = (int)(((nelem >> 16) & 0xFFFFu) | (1u << 16));        // dim0[31:16] | tensor_dim1=1
  g1[3] = (int)((nelem & 0xFFFFu) << 16);                       // tile_dim0
  g1[4] = 1;                                                    // tile_dim1=1, tile_dim2=0
  g1[5] = (int)nelem;                                           // tensor_dim0_stride[31:0]
  g1[6] = 0;
  g1[7] = 0;
  v4i_ z4 = {0, 0, 0, 0};
  v8i_ z8 = {0, 0, 0, 0, 0, 0, 0, 0};
  __builtin_amdgcn_tensor_load_to_lds(g0, g1, z4, z4, z8, 0);
}
#endif

// ---------------- prep kernels ----------------
__global__ void cvt_bf16_kernel(const float* __restrict__ src,
                                __bf16* __restrict__ dst, int n) {
  int i = blockIdx.x * blockDim.x + threadIdx.x;
  if (i < n) dst[i] = (__bf16)src[i];
}

__global__ void embed_kernel(const int* __restrict__ tokens,       // [B][S]
                             const float* __restrict__ table,      // [V][E]
                             __bf16* __restrict__ out) {           // [S][B][E]
  int s = blockIdx.x;
  int b = blockIdx.y;
  int e = threadIdx.x;                           // 256 threads
  int tok = tokens[b * SEQN + s];
  out[((size_t)s * BATCH + b) * EMBD + e] = (__bf16)table[(size_t)tok * EMBD + e];
}

__global__ void dest_kernel(const int* __restrict__ tokens,        // [B][S]
                            int* __restrict__ dest) {              // [S][B]
  int b = threadIdx.x;                           // 64 threads
  int cnt = 0;
  for (int s = 0; s < SEQN; ++s) {
    int tok = tokens[b * SEQN + s];
    int d = -1;
    if (tok == EOS) d = cnt++;
    dest[s * BATCH + b] = d;
  }
}

// ---------------- persistent GRU kernel ----------------
__launch_bounds__(NT, 1)
__global__ void gru_persistent(const float* __restrict__ bih,
                               const float* __restrict__ bhh,
                               const __bf16* __restrict__ wih_bf,   // [1536][256]
                               const __bf16* __restrict__ whh_bf,   // [1536][512]
                               const __bf16* __restrict__ emb_bf,   // [S][64][256]
                               const int*   __restrict__ dest,      // [S][64]
                               __bf16* __restrict__ hbf,            // 2 x [64][512] bf16 ping-pong
                               unsigned* __restrict__ bar,
                               float* __restrict__ out) {           // [32][64][512]
  __shared__ float lds_gi[BATCH * 96];
  __shared__ float lds_gh[BATCH * 96];
  __shared__ float lds_bi[96];
  __shared__ float lds_bh[96];
  __shared__ float lds_hprev[BATCH * HS];         // own 64x32 f32 slice, persistent
  extern __shared__ char dynlds[];                // h slab + 2 emb slabs (padded rows)

  const int wg   = blockIdx.x;            // 0..15
  const int j0   = wg * HS;               // hidden column base of this WG
  const int tid  = threadIdx.x;
  const int wave = tid >> 5;              // 0..11
  const int lane = tid & 31;
  const int ni   = wave >> 1;             // 0..5  : 16 gate columns
  const int mih  = wave & 1;              // tile rows: mih*32 .. mih*32+31
  const int m0a  = mih * 32;
  const int ncol0 = ni * 16;              // local gate column base
  const int group = ncol0 >> 5;           // 0=r, 1=z, 2=n
  const int gcol_base = group * HIDN + j0 + (ncol0 & 31);

  const unsigned ldsH  = (unsigned)(unsigned long long)(void*)dynlds;
  const unsigned ldsE0 = ldsH + (unsigned)HLDS_BYTES;

  // biases for this WG's 96 local gate columns -> LDS; init persistent h slice
  for (int c = tid; c < 96; c += NT) {
    int gcol = (c >> 5) * HIDN + j0 + (c & 31);
    lds_bi[c] = bih[gcol];
    lds_bh[c] = bhh[gcol];
  }
  for (int idx = tid; idx < BATCH * HS; idx += NT) lds_hprev[idx] = 0.0f;

  // per-lane fragment addressing
  const int brow  = gcol_base + (lane & 15);        // weight row (gate col)
  const int bkup  = (lane >> 4) << 3;               // +8 K for upper half-wave (B layout)
  const int mrow  = lane & 15;                      // A row within tile
  const int akoff = (lane >> 4) << 3;               // +8 K for upper half-wave (A layout)

  // ALL B fragments register resident for the whole sequence (24 x 8 VGPRs = 192)
  v16bf Bi[8], Bh[16];
  {
    const __bf16* biPtr = wih_bf + (size_t)brow * EMBD + bkup;
    const __bf16* bhPtr = whh_bf + (size_t)brow * HIDN + bkup;
#pragma unroll
    for (int k = 0; k < 8; ++k)  Bi[k] = load_frag_split(biPtr + k * 32);
#pragma unroll
    for (int k = 0; k < 16; ++k) Bh[k] = load_frag_split(bhPtr + k * 32);
  }

  // prologue: emb slab for t=0 into LDS buffer 0
#if HAVE_TDM
  if (wave == 0) {
    tdm_load_1d(emb_bf, ldsE0, BATCH * EMBD, /*512B rows*/ 5, /*4 DW pad*/ 3);
    __builtin_amdgcn_s_wait_tensorcnt(0);
  }
#else
  {
    const uint4* esrc = (const uint4*)emb_bf;
    uint4* edst = (uint4*)(dynlds + HLDS_BYTES);
#pragma unroll
    for (int i = 0; i < 6; ++i) {
      int idx = tid + i * NT;
      if (idx < BATCH * 32) edst[(idx >> 5) * EROW_U4 + (idx & 31)] = esrc[idx];
    }
  }
#endif
  __syncthreads();

  int p = 0;

  for (int t = 0; t < SEQN; ++t) {
    const size_t hoff_p = (size_t)p * HSTRIDE;
    const size_t hoff_n = (size_t)(p ^ 1) * HSTRIDE;

    // ---- full h (bf16, 64KB) global -> padded LDS slab ----
#if HAVE_TDM
    if (wave == 0) {
      tdm_load_1d(hbf + hoff_p, ldsH, BATCH * HIDN, /*1024B rows*/ 7, /*4 DW pad*/ 3);
      __builtin_amdgcn_s_wait_tensorcnt(0);
    }
#else
    {
      const uint4* hsrc = (const uint4*)(hbf + hoff_p);
      uint4* h_u4 = (uint4*)dynlds;
#pragma unroll
      for (int i = 0; i < 11; ++i) {
        int idx = tid + i * NT;
        if (idx < BATCH * 64) h_u4[(idx >> 6) * HROW_U4 + (idx & 63)] = hsrc[idx];
      }
    }
#endif
    __syncthreads();

    const __bf16* eA = (const __bf16*)(dynlds + HLDS_BYTES + (size_t)(t & 1) * ELDS_BYTES);
    const __bf16* hA = (const __bf16*)dynlds;

    // A-fragment from LDS for macro K-iteration i (0..7: emb, 8..23: h)
    auto loadA = [&](int i, int tt) -> v16bf {
      const int row = m0a + tt * 16 + mrow;
      if (i < 8)
        return load_frag_split(eA + (size_t)row * EROW_E + i * 32 + akoff);
      else
        return load_frag_split(hA + (size_t)row * HROW_E + (i - 8) * 32 + akoff);
    };

    v8f ai[2] = {vzero8(), vzero8()};
    v8f ah[2] = {vzero8(), vzero8()};

    v16bf stage[PF][2];
#pragma unroll
    for (int i = 0; i < PF; ++i) {
      stage[i][0] = loadA(i, 0);
      stage[i][1] = loadA(i, 1);
    }
#pragma unroll
    for (int i = 0; i < 24; ++i) {
      const int s = i % PF;
      v16bf a0 = stage[s][0];
      v16bf a1 = stage[s][1];
      if (i + PF < 24) {
        stage[s][0] = loadA(i + PF, 0);
        stage[s][1] = loadA(i + PF, 1);
      }
      if (i < 8) {
        ai[0] = __builtin_amdgcn_wmma_f32_16x16x32_bf16(false, a0, false, Bi[i], (short)0, ai[0], false, false);
        ai[1] = __builtin_amdgcn_wmma_f32_16x16x32_bf16(false, a1, false, Bi[i], (short)0, ai[1], false, false);
      } else {
        ah[0] = __builtin_amdgcn_wmma_f32_16x16x32_bf16(false, a0, false, Bh[i - 8], (short)0, ah[0], false, false);
        ah[1] = __builtin_amdgcn_wmma_f32_16x16x32_bf16(false, a1, false, Bh[i - 8], (short)0, ah[1], false, false);
      }
    }

    // ---- next step's emb slab into the other LDS buffer (no barrier dep) ----
    if (t + 1 < SEQN) {
#if HAVE_TDM
      if (wave == 1) {
        tdm_load_1d(emb_bf + (size_t)(t + 1) * (BATCH * EMBD),
                    ldsE0 + (unsigned)(((t + 1) & 1) * ELDS_BYTES),
                    BATCH * EMBD, 5, 3);
      }
#else
      const uint4* esrc = (const uint4*)(emb_bf + (size_t)(t + 1) * (BATCH * EMBD));
      uint4* edst = (uint4*)(dynlds + HLDS_BYTES + (size_t)((t + 1) & 1) * ELDS_BYTES);
#pragma unroll
      for (int i = 0; i < 6; ++i) {
        int idx = tid + i * NT;
        if (idx < BATCH * 32) edst[(idx >> 5) * EROW_U4 + (idx & 31)] = esrc[idx];
      }
#endif
    }

    // write accumulators to LDS (ISA f32 C/D layout: VGPR v, lane l -> M=v+8*(l>=16), N=l&15)
    {
      const int nloc = ncol0 + (lane & 15);
      const int mtop = (lane >> 4) << 3;
#pragma unroll
      for (int tt = 0; tt < 2; ++tt) {
        const int mb = m0a + tt * 16 + mtop;
#pragma unroll
        for (int v = 0; v < 8; ++v) {
          lds_gi[(mb + v) * 96 + nloc] = ai[tt][v];
          lds_gh[(mb + v) * 96 + nloc] = ah[tt][v];
        }
      }
    }
    __syncthreads();

    // ---- gate math for this WG's 64 x 32 hidden slice ----
    const int* dst_t = dest + t * BATCH;
    for (int idx = tid; idx < BATCH * HS; idx += NT) {
      int b = idx >> 5;
      int j = idx & 31;
      float gir = lds_gi[b * 96 + j]      + lds_bi[j];
      float ghr = lds_gh[b * 96 + j]      + lds_bh[j];
      float giz = lds_gi[b * 96 + 32 + j] + lds_bi[32 + j];
      float ghz = lds_gh[b * 96 + 32 + j] + lds_bh[32 + j];
      float gin = lds_gi[b * 96 + 64 + j] + lds_bi[64 + j];
      float ghn = lds_gh[b * 96 + 64 + j] + lds_bh[64 + j];
      float r = fast_sigmoid(gir + ghr);
      float z = fast_sigmoid(giz + ghz);
      float n = fast_tanh(gin + r * ghn);
      float hp = lds_hprev[idx];                       // own slice, LDS-persistent f32
      float hn = (1.0f - z) * n + z * hp;
      lds_hprev[idx] = hn;
      hbf[hoff_n + b * HIDN + j0 + j] = (__bf16)hn;    // publish bf16 h for all WGs
      int d = dst_t[b];
      if (d >= 0) out[((size_t)d * BATCH + b) * HIDN + j0 + j] = hn;
    }

#if HAVE_TDM
    if (wave == 1) __builtin_amdgcn_s_wait_tensorcnt(0);   // emb(t+1) landed before barrier
#endif

    // ---- grid barrier (ping-pong buffers => one barrier per step suffices) ----
    __syncthreads();
    if (tid == 0) {
      __threadfence();
      unsigned g = __atomic_load_n(&bar[1], __ATOMIC_ACQUIRE);
      unsigned prev = atomicAdd(&bar[0], 1u);
      if (prev == NWG - 1) {
        __atomic_store_n(&bar[0], 0u, __ATOMIC_RELAXED);
        __threadfence();
        atomicAdd(&bar[1], 1u);
      } else {
        while (__atomic_load_n(&bar[1], __ATOMIC_ACQUIRE) == g) {
          __builtin_amdgcn_s_sleep(1);
        }
      }
      __threadfence();
    }
    __syncthreads();

    p ^= 1;
  }
}

// ---------------- host entry ----------------
extern "C" void kernel_launch(void* const* d_in, const int* in_sizes, int n_in,
                              void* d_out, int out_size, void* d_ws, size_t ws_size,
                              hipStream_t stream) {
  const int*   tokens = (const int*)d_in[0];
  const float* table  = (const float*)d_in[1];
  const float* wih    = (const float*)d_in[2];
  const float* whh    = (const float*)d_in[3];
  const float* bih    = (const float*)d_in[4];
  const float* bhh    = (const float*)d_in[5];
  float* out = (float*)d_out;
  char* ws = (char*)d_ws;

  unsigned* bar   = (unsigned*)(ws + OFF_BAR);
  __bf16*   hbf   = (__bf16*)(ws + OFF_HBF);
  int*      dest  = (int*)(ws + OFF_DEST);
  __bf16*   wihbf = (__bf16*)(ws + OFF_WIH);
  __bf16*   whhbf = (__bf16*)(ws + OFF_WHH);
  __bf16*   embbf = (__bf16*)(ws + OFF_EMBB);

  // zero: barrier + both bf16 h ping-pong buffers (h0 = 0); graph-capture safe
  (void)hipMemsetAsync(ws, 0, OFF_DEST, stream);

  cvt_bf16_kernel<<<(GATE3 * EMBD + 255) / 256, 256, 0, stream>>>(wih, wihbf, GATE3 * EMBD);
  cvt_bf16_kernel<<<(GATE3 * HIDN + 255) / 256, 256, 0, stream>>>(whh, whhbf, GATE3 * HIDN);
  embed_kernel<<<dim3(SEQN, BATCH), EMBD, 0, stream>>>(tokens, table, embbf);
  dest_kernel<<<1, BATCH, 0, stream>>>(tokens, dest);

  gru_persistent<<<NWG, NT, DYN_BYTES, stream>>>(bih, bhh, wihbf, whhbf, embbf, dest,
                                                 hbf, bar, out);
}
